// TDGNN_3633542332753
// MI455X (gfx1250) — matrix operand order repaired and checked by reading, not compiled, not run on hardware
//
#include <hip/hip_runtime.h>
#include <hip/hip_bf16.h>

typedef __attribute__((ext_vector_type(16))) _Float16 v16h;
typedef __attribute__((ext_vector_type(8)))  float    v8f;

#define DF 128
#define DH 256

// ---------------------------------------------------------------------------
// Pack an f32 weight matrix w[nk, ncols] into f16 WMMA B-fragment order:
//   pw[((tile*kiters + kit)*32 + lane)*16 + j]
// with lane = (bit 3 of k within 32-K slab)*16 + n, j = (bit 4 of k)*8 + (k&7)  (B striping),
// tile = column-tile of 16, kit = K-slab of 32.
// ---------------------------------------------------------------------------
__global__ __launch_bounds__(256)
void tdgnn_pack_w(const float* __restrict__ w, _Float16* __restrict__ pw,
                  int ncols, int nk)
{
    const int idx   = blockIdx.x * 256 + threadIdx.x;
    const int total = nk * ncols;
    if (idx >= total) return;
    const int j      = idx & 15;
    const int lane   = (idx >> 4) & 31;
    const int frag   = idx >> 9;
    const int kiters = nk >> 5;
    const int tile   = frag / kiters;
    const int kit    = frag - tile * kiters;
    const int half   = lane >> 4;
    const int n      = lane & 15;
    const int k      = (kit << 5) + ((j >> 3) << 4) + (half << 3) + (j & 7);
    pw[idx] = (_Float16)w[(size_t)k * ncols + (tile << 4) + n];
}

// ---------------------------------------------------------------------------
// Fused MLP: out = relu(X @ W1 + b1) @ W2 + b2 per 16-row tile, plus
// acc = prop_w[0]*out.  A operands live in LDS in fragment order (one
// contiguous 32B read per WMMA); B operands are pre-packed f16 fragments
// (two global_load_b128 per WMMA).  128 threads = 4 waves (wave32).
// ---------------------------------------------------------------------------
__global__ __launch_bounds__(128)
void tdgnn_mlp_wmma(const float* __restrict__ x,
                    const _Float16* __restrict__ pw1, const float* __restrict__ b1,
                    const _Float16* __restrict__ pw2, const float* __restrict__ b2,
                    const float* __restrict__ prop_w,
                    float* __restrict__ h_out, float* __restrict__ acc_out,
                    int n_nodes)
{
    __shared__ __align__(32) _Float16 sXf[4][32][16];   // GEMM1 A fragments, 4 KB
    __shared__ __align__(32) _Float16 sHf[8][32][16];   // GEMM2 A fragments, 8 KB

    const int row0 = blockIdx.x * 16;
    const int tid  = threadIdx.x;
    const int wid  = tid >> 5;
    const int lane = tid & 31;
    const int m    = lane & 15;
    const int half = lane >> 4;
    const float pw0 = prop_w[0];

    // Stage X tile -> fragment-ordered LDS (ISA 16-bit A 16x32 layout:
    // K = (v<4?0:16) + half*8 + (v&3)*2 + i  =>  inverse below).
    for (int i = tid; i < 16 * DF; i += 128) {
        const int r = i >> 7;
        const int k = i & 127;
        int rr = row0 + r; if (rr >= n_nodes) rr = n_nodes - 1;
        const float val = x[(size_t)rr * DF + k];
        const int kit = k >> 5, kk = k & 31;
        const int q = kk >> 1, ib = kk & 1;
        const int hf = (q >> 2) & 1;
        const int v  = (q & 3) + ((q >= 8) ? 4 : 0);
        sXf[kit][hf * 16 + r][2 * v + ib] = (_Float16)val;
    }
    __syncthreads();

    // ---- GEMM1: H[16,256] = relu(X @ W1 + b1); 4 col-tiles per wave ----
    for (int t = 0; t < 4; ++t) {
        const int tile = wid * 4 + t;
        const int n0   = tile * 16;
        v8f c = {};
        for (int kit = 0; kit < 4; ++kit) {
            const v16h a = *(const v16h*)&sXf[kit][lane][0];
            const v16h b = *(const v16h*)&pw1[(size_t)((tile * 4 + kit) * 32 + lane) * 16];
            c = __builtin_amdgcn_wmma_f32_16x16x32_f16(false, a, false, b,
                                                       (short)0, c, false, false);
        }
        const int   cc   = n0 + m;          // output col = K index for GEMM2
        const float bias = b1[cc];
        // Inverse A-layout mapping for K=cc (fixed per lane):
        const int kit2 = cc >> 5, kk2 = cc & 31;
        const int q2 = kk2 >> 1, i2 = kk2 & 1;
        const int hf2 = (q2 >> 2) & 1;
        const int v2  = (q2 & 3) + ((q2 >= 8) ? 4 : 0);
        const int j2  = 2 * v2 + i2;
#pragma unroll
        for (int v = 0; v < 8; ++v) {
            const int rr = v + half * 8;     // C layout: M = v + 8*half
            sHf[kit2][hf2 * 16 + rr][j2] = (_Float16)fmaxf(c[v] + bias, 0.0f);
        }
    }
    __syncthreads();

    // ---- GEMM2: OUT[16,128] = H @ W2 + b2; 2 col-tiles per wave ----
    for (int t = 0; t < 2; ++t) {
        const int tile = wid * 2 + t;
        const int n0   = tile * 16;
        v8f c = {};
        for (int kit = 0; kit < 8; ++kit) {
            const v16h a = *(const v16h*)&sHf[kit][lane][0];
            const v16h b = *(const v16h*)&pw2[(size_t)((tile * 8 + kit) * 32 + lane) * 16];
            c = __builtin_amdgcn_wmma_f32_16x16x32_f16(false, a, false, b,
                                                       (short)0, c, false, false);
        }
        const int   cc   = n0 + m;
        const float bias = b2[cc];
#pragma unroll
        for (int v = 0; v < 8; ++v) {
            const int rr = row0 + v + half * 8;
            if (rr < n_nodes) {
                const float val = c[v] + bias;
                h_out[(size_t)rr * DF + cc]   = val;
                acc_out[(size_t)rr * DF + cc] = pw0 * val;
            }
        }
    }
}

// ---------------------------------------------------------------------------
// Weighted scatter for one hop (blockIdx.y = hop):
//   acc[dst] += prop_w[hop+1] * ew[hop][e] * h[src]
// One wave per edge; each lane covers float4 of the 128 features.
// h + acc (102 MB) are L2-resident (192 MB L2): gathers + f32 atomics stay
// in L2; HBM only streams the edge lists.
// ---------------------------------------------------------------------------
__global__ __launch_bounds__(256)
void tdgnn_scatter(const float* __restrict__ h,
                   const int* __restrict__ edge_index,
                   const float* __restrict__ edge_weight,
                   const float* __restrict__ prop_w,
                   float* __restrict__ acc, int n_edges)
{
    const int hop = blockIdx.y;
    const long long gtid = (long long)blockIdx.x * blockDim.x + threadIdx.x;
    const int e    = (int)(gtid >> 5);
    const int lane = (int)(gtid & 31);
    if (e >= n_edges) return;

    const size_t base = (size_t)hop * 2 * n_edges;
    const int src = edge_index[base + e];
    const int dst = edge_index[base + n_edges + e];
    const float w = edge_weight[(size_t)hop * n_edges + e] * prop_w[hop + 1];

    const int f = lane * 4;
    const float4 v = *(const float4*)(h + (size_t)src * DF + f);
    float* p = acc + (size_t)dst * DF + f;
    atomicAdd(p + 0, w * v.x);
    atomicAdd(p + 1, w * v.y);
    atomicAdd(p + 2, w * v.z);
    atomicAdd(p + 3, w * v.w);
}

__global__ __launch_bounds__(256)
void tdgnn_zero(float* __restrict__ p, int n)
{
    const int i = blockIdx.x * 256 + threadIdx.x;
    if (i < n) p[i] = 0.0f;
}

// pooled[batch[n]] += acc[n]; one wave per node, float4 per lane.
__global__ __launch_bounds__(256)
void tdgnn_pool(const float* __restrict__ acc, const int* __restrict__ batch,
                float* __restrict__ pooled, int n_nodes)
{
    const long long gtid = (long long)blockIdx.x * blockDim.x + threadIdx.x;
    const int node = (int)(gtid >> 5);
    const int lane = (int)(gtid & 31);
    if (node >= n_nodes) return;
    const int g = batch[node];
    const int f = lane * 4;
    const float4 v = *(const float4*)(acc + (size_t)node * DF + f);
    float* p = pooled + (size_t)g * DF + f;
    atomicAdd(p + 0, v.x);
    atomicAdd(p + 1, v.y);
    atomicAdd(p + 2, v.z);
    atomicAdd(p + 3, v.w);
}

// Row-wise log_softmax over [G, 128]; one 128-thread block per graph.
__global__ __launch_bounds__(128)
void tdgnn_logsoftmax(const float* __restrict__ pooled, float* __restrict__ out)
{
    __shared__ float red[128];
    const int g = blockIdx.x;
    const int t = threadIdx.x;
    const float v = pooled[(size_t)g * DF + t];

    red[t] = v;
    __syncthreads();
    for (int s = 64; s > 0; s >>= 1) {
        if (t < s) red[t] = fmaxf(red[t], red[t + s]);
        __syncthreads();
    }
    const float mx = red[0];
    __syncthreads();

    red[t] = __expf(v - mx);
    __syncthreads();
    for (int s = 64; s > 0; s >>= 1) {
        if (t < s) red[t] += red[t + s];
        __syncthreads();
    }
    const float lse = __logf(red[0]) + mx;
    out[(size_t)g * DF + t] = v - lse;
}

// ---------------------------------------------------------------------------
extern "C" void kernel_launch(void* const* d_in, const int* in_sizes, int n_in,
                              void* d_out, int out_size, void* d_ws, size_t ws_size,
                              hipStream_t stream)
{
    const float* x           = (const float*)d_in[0];
    const int*   edge_index  = (const int*)  d_in[1];
    const float* edge_weight = (const float*)d_in[2];
    const int*   batch       = (const int*)  d_in[3];
    const float* w1          = (const float*)d_in[4];
    const float* b1          = (const float*)d_in[5];
    const float* w2          = (const float*)d_in[6];
    const float* b2          = (const float*)d_in[7];
    const float* prop_w      = (const float*)d_in[8];
    float*       out         = (float*)d_out;

    const int N = in_sizes[0] / DF;   // 100000 nodes
    const int E = in_sizes[2] / 3;    // 1600000 edges per hop
    const int G = out_size / DF;      // 64 graphs

    float*     ws_h      = (float*)d_ws;
    float*     ws_acc    = ws_h   + (size_t)N * DF;
    float*     ws_pooled = ws_acc + (size_t)N * DF;
    _Float16*  ws_pw1    = (_Float16*)(ws_pooled + (size_t)G * DF);
    _Float16*  ws_pw2    = ws_pw1 + (size_t)DF * DH;

    // 0) Pack weights to f16 WMMA B-fragment order (tiny, L2-resident)
    tdgnn_pack_w<<<(DF * DH + 255) / 256, 256, 0, stream>>>(w1, ws_pw1, DH, DF);
    tdgnn_pack_w<<<(DH * DF + 255) / 256, 256, 0, stream>>>(w2, ws_pw2, DF, DH);

    // 1) Fused MLP (WMMA) -> h, and acc = prop_w[0]*h
    const int nblk = (N + 15) / 16;
    tdgnn_mlp_wmma<<<nblk, 128, 0, stream>>>(x, ws_pw1, b1, ws_pw2, b2, prop_w,
                                             ws_h, ws_acc, N);

    // 2) Three hops of weighted scatter-add (hop = blockIdx.y)
    const long long hthreads = (long long)E * 32;
    dim3 sgrid((unsigned)((hthreads + 255) / 256), 3, 1);
    tdgnn_scatter<<<sgrid, 256, 0, stream>>>(ws_h, edge_index, edge_weight,
                                             prop_w, ws_acc, E);

    // 3) Global add-pool
    tdgnn_zero<<<(G * DF + 255) / 256, 256, 0, stream>>>(ws_pooled, G * DF);
    const long long pthreads = (long long)N * 32;
    tdgnn_pool<<<(int)((pthreads + 255) / 256), 256, 0, stream>>>(
        ws_acc, batch, ws_pooled, N);

    // 4) log_softmax per graph
    tdgnn_logsoftmax<<<G, 128, 0, stream>>>(ws_pooled, out);
}
